// LocalDeliberationBlock_65266323030409
// MI455X (gfx1250) — compile-verified, matrix operand order checked by reading, not run on hardware
//
#include <hip/hip_runtime.h>
#include <hip/hip_bf16.h>
#include <math.h>

// ---------------------------------------------------------------------------
// LocalDeliberationBlock for MI455X (gfx1250, wave32, WMMA).
// All dense GEMMs run on v_wmma_f32_16x16x32_bf16 (f32 accumulate).
// Weights are pre-packed into WMMA B-fragment layout; activations are loaded
// as A fragments directly from row-major bf16 (2 x b128 per lane).
// GEMM waves compute 32x64 tiles with an explicit double-buffered K pipeline.
// ---------------------------------------------------------------------------

#define BATCH   4
#define SEQ     2048
#define DMODEL  1024
#define DSTATE  256
#define NTOK    (BATCH * SEQ)          // 8192
#define KSIZE   5
#define CHUNK   16
#define STEPS   2
#define TOPK    8
#define LOOKB   64
#define NCHUNKR (NTOK / CHUNK)         // 512 global chunk rows

typedef __attribute__((ext_vector_type(16))) __bf16        v16bf;
typedef __attribute__((ext_vector_type(8)))  float         v8f;
typedef __attribute__((ext_vector_type(4)))  unsigned int  v4u;

union Frag { v16bf v; v4u u[2]; };

__device__ __forceinline__ __bf16 f2bf(float f) {
    union { float f; unsigned u; } x; x.f = f;
    unsigned r = x.u + 0x7FFFu + ((x.u >> 16) & 1u);   // RNE
    unsigned short h = (unsigned short)(r >> 16);
    return __builtin_bit_cast(__bf16, h);
}

#define NEG_INF (-__builtin_inff())

// ---------------------------------------------------------------------------
// Weight packing: W (f32, row-major [K,N]) -> bf16 WMMA-B fragment layout.
// packed[((kt*ntN + nt)*32 + lane)*16 + e] = W[kt*32 + kmap(lane,e)][nt*16 + lane%16]
// kmap(lane,e) = (e<8 ? e : e+8) + 8*(lane/16)   (mirrors 16-bit A K packing)
// ---------------------------------------------------------------------------
__global__ void pack_w_kernel(const float* __restrict__ W, int K, int N,
                              __bf16* __restrict__ out) {
    int idx = blockIdx.x * blockDim.x + threadIdx.x;
    if (idx >= K * N) return;
    int e    = idx & 15;
    int lane = (idx >> 4) & 31;
    int ntN  = N >> 4;
    int nt   = (idx >> 9) % ntN;
    int kt   = idx / (ntN << 9);
    int kk   = ((e < 8) ? e : e + 8) + ((lane >> 4) << 3);
    int krow = (kt << 5) + kk;
    int ncol = (nt << 4) + (lane & 15);
    out[idx] = f2bf(W[(size_t)krow * N + ncol]);
}

__global__ void convert_bf16_kernel(const float* __restrict__ in,
                                    __bf16* __restrict__ out, int n) {
    int idx = blockIdx.x * blockDim.x + threadIdx.x;
    if (idx < n) out[idx] = f2bf(in[idx]);
}

// ---------------------------------------------------------------------------
// GEMM: C[M,N] = act(A_bf16[M,K] @ Bpacked + bias) (+ residual)
// 8 waves/block; each wave computes a 32x64 tile (2 M-subtiles x 4 N-tiles,
// 8 WMMAs per 32-deep K step, B fragments shared across M-subtiles).
// Double-buffered K pipeline: next tile's fragments load while current MMAs
// issue, so s_wait_loadcnt overlaps the WMMA block. act: 0=none, 1=tanh.
// ---------------------------------------------------------------------------
__global__ void gemm_bf16_wmma_kernel(const __bf16* __restrict__ A,
                                      const __bf16* __restrict__ Bp,
                                      const float*  __restrict__ bias,
                                      float*        __restrict__ Cf, int ldcf,
                                      __bf16*       __restrict__ Cb, int ldcb,
                                      const float*  __restrict__ res, int ldr,
                                      int M, int N, int K, int act) {
    const int wave  = threadIdx.x >> 5;
    const int lane  = threadIdx.x & 31;
    const int tileM = blockIdx.x * 256 + wave * 32;
    const int ntBase = blockIdx.y * 4;
    if (tileM >= M) return;

    const int half = lane >> 4;          // K-half owned by this lane
    const int row0 = tileM + (lane & 15);
    const int ntN  = N >> 4;
    const int kTiles = K >> 5;

    const __bf16* a0row = A + (size_t)row0 * K + (half << 3);
    const __bf16* a1row = a0row + (size_t)16 * K;

    v8f acc[8] = {};
    Frag a0c, a1c, bc0, bc1, bc2, bc3;
    Frag a0n, a1n, bn0, bn1, bn2, bn3;

#define LOAD_TILE(kt, A0, A1, B0, B1, B2, B3)                                  \
    do {                                                                       \
        const __bf16* ap0 = a0row + ((kt) << 5);                               \
        const __bf16* ap1 = a1row + ((kt) << 5);                               \
        (A0).u[0] = *(const v4u*)ap0;  (A0).u[1] = *(const v4u*)(ap0 + 16);    \
        (A1).u[0] = *(const v4u*)ap1;  (A1).u[1] = *(const v4u*)(ap1 + 16);    \
        __builtin_prefetch(ap0 + 32, 0, 1);                                    \
        const __bf16* bp = Bp + (((size_t)(kt) * ntN + ntBase) << 9)           \
                              + ((size_t)lane << 4);                           \
        (B0).u[0] = *(const v4u*)bp;            (B0).u[1] = *(const v4u*)(bp + 8);      \
        (B1).u[0] = *(const v4u*)(bp + 512);    (B1).u[1] = *(const v4u*)(bp + 520);    \
        (B2).u[0] = *(const v4u*)(bp + 1024);   (B2).u[1] = *(const v4u*)(bp + 1032);   \
        (B3).u[0] = *(const v4u*)(bp + 1536);   (B3).u[1] = *(const v4u*)(bp + 1544);   \
    } while (0)

#define MMA_TILE(A0, A1, B0, B1, B2, B3)                                       \
    do {                                                                       \
        acc[0] = __builtin_amdgcn_wmma_f32_16x16x32_bf16(false, (A0).v, false, (B0).v, (short)0, acc[0], false, false); \
        acc[1] = __builtin_amdgcn_wmma_f32_16x16x32_bf16(false, (A0).v, false, (B1).v, (short)0, acc[1], false, false); \
        acc[2] = __builtin_amdgcn_wmma_f32_16x16x32_bf16(false, (A0).v, false, (B2).v, (short)0, acc[2], false, false); \
        acc[3] = __builtin_amdgcn_wmma_f32_16x16x32_bf16(false, (A0).v, false, (B3).v, (short)0, acc[3], false, false); \
        acc[4] = __builtin_amdgcn_wmma_f32_16x16x32_bf16(false, (A1).v, false, (B0).v, (short)0, acc[4], false, false); \
        acc[5] = __builtin_amdgcn_wmma_f32_16x16x32_bf16(false, (A1).v, false, (B1).v, (short)0, acc[5], false, false); \
        acc[6] = __builtin_amdgcn_wmma_f32_16x16x32_bf16(false, (A1).v, false, (B2).v, (short)0, acc[6], false, false); \
        acc[7] = __builtin_amdgcn_wmma_f32_16x16x32_bf16(false, (A1).v, false, (B3).v, (short)0, acc[7], false, false); \
    } while (0)

    LOAD_TILE(0, a0c, a1c, bc0, bc1, bc2, bc3);
    for (int kt = 0; kt < kTiles - 1; ++kt) {
        LOAD_TILE(kt + 1, a0n, a1n, bn0, bn1, bn2, bn3);
        MMA_TILE(a0c, a1c, bc0, bc1, bc2, bc3);
        a0c = a0n; a1c = a1n;
        bc0 = bn0; bc1 = bn1; bc2 = bn2; bc3 = bn3;
    }
    MMA_TILE(a0c, a1c, bc0, bc1, bc2, bc3);

#undef LOAD_TILE
#undef MMA_TILE

    const int colBase = (ntBase << 4) + (lane & 15);
#pragma unroll
    for (int mt = 0; mt < 2; ++mt) {
#pragma unroll
        for (int j = 0; j < 4; ++j) {
            const int col = colBase + (j << 4);
            const float bv = bias ? bias[col] : 0.f;
#pragma unroll
            for (int r = 0; r < 8; ++r) {
                const int orow = tileM + mt * 16 + r + (half << 3);
                float xv = acc[mt * 4 + j][r] + bv;
                if (act == 1) xv = tanhf(xv);
                if (res) xv += res[(size_t)orow * ldr + col];
                if (Cf)  Cf[(size_t)orow * ldcf + col] = xv;
                if (Cb)  Cb[(size_t)orow * ldcb + col] = f2bf(xv);
            }
        }
    }
}

// ---------------------------------------------------------------------------
// Seed: h_bf16 and ui block 0 from freshly computed h (f32).
// ---------------------------------------------------------------------------
__global__ void seed_h_kernel(const float* __restrict__ h,
                              __bf16* __restrict__ h_bf,
                              __bf16* __restrict__ ui) {
    int idx = blockIdx.x * blockDim.x + threadIdx.x;   // NTOK*DSTATE grid
    int d = idx & (DSTATE - 1);
    int t = idx >> 8;
    __bf16 b = f2bf(h[idx]);
    h_bf[idx] = b;
    ui[(size_t)t * DMODEL + d] = b;
}

// ---------------------------------------------------------------------------
// Causal depthwise conv (K=5) -> ui block 1 (bf16).
// ---------------------------------------------------------------------------
__global__ void conv_kernel(const float* __restrict__ h,
                            const float* __restrict__ conv_w,
                            const float* __restrict__ conv_b,
                            __bf16* __restrict__ ui) {
    int idx = blockIdx.x * blockDim.x + threadIdx.x;
    int d = idx & (DSTATE - 1);
    int t = idx >> 8;
    int s = t & (SEQ - 1);
    float acc = conv_b[d];
#pragma unroll
    for (int kk = 0; kk < KSIZE; ++kk) {
        int sp = s - (KSIZE - 1) + kk;
        float hv = (sp >= 0) ? h[(size_t)(t - (KSIZE - 1) + kk) * DSTATE + d] : 0.f;
        acc += hv * conv_w[kk * DSTATE + d];
    }
    ui[(size_t)t * DMODEL + DSTATE + d] = f2bf(acc);
}

// ---------------------------------------------------------------------------
// Chunk means of h -> ph (bf16, GEMM A operand).
// ---------------------------------------------------------------------------
__global__ void chunk_mean_kernel(const float* __restrict__ h,
                                  __bf16* __restrict__ ph_bf) {
    int idx = blockIdx.x * blockDim.x + threadIdx.x;   // NCHUNKR*DSTATE
    int d = idx & (DSTATE - 1);
    int row = idx >> 8;
    const float* base = h + (size_t)row * CHUNK * DSTATE + d;
    float s = 0.f;
#pragma unroll
    for (int i = 0; i < CHUNK; ++i) s += base[(size_t)i * DSTATE];
    ph_bf[idx] = f2bf(s * (1.f / CHUNK));
}

// pb (bf16, per chunk row) broadcast into ui block 2.
__global__ void broadcast_pb_kernel(const __bf16* __restrict__ pb_bf,
                                    __bf16* __restrict__ ui) {
    int idx = blockIdx.x * blockDim.x + threadIdx.x;   // NTOK*DSTATE
    int d = idx & (DSTATE - 1);
    int t = idx >> 8;
    ui[(size_t)t * DMODEL + 2 * DSTATE + d] = pb_bf[(size_t)(t >> 4) * DSTATE + d];
}

// ---------------------------------------------------------------------------
// Windowed causal top-8 attention. One wave32 per token.
// ---------------------------------------------------------------------------
__global__ void attn_topk_kernel(const float* __restrict__ q,
                                 const float* __restrict__ k,
                                 const float* __restrict__ v,
                                 __bf16* __restrict__ ui) {
    __shared__ __align__(16) float qs[8][DSTATE];
    const int wave = threadIdx.x >> 5;
    const int lane = threadIdx.x & 31;
    const int t = blockIdx.x * 8 + wave;

    // stage q[t] into LDS (64 float4)
    const float4* qrow = (const float4*)(q + (size_t)t * DSTATE);
    float4* qsv = (float4*)qs[wave];
    qsv[lane]      = qrow[lane];
    qsv[lane + 32] = qrow[lane + 32];
    __syncthreads();

    const int s_in_b = t & (SEQ - 1);
    const int bstart = t - s_in_b;

    // scores for 2 candidates per lane (rel index r in [0,64))
    float sc[2];
#pragma unroll
    for (int c = 0; c < 2; ++c) {
        const int r = 2 * lane + c;
        const int rel = s_in_b - LOOKB + r;
        if (rel < 0) { sc[c] = NEG_INF; continue; }
        const float4* krow = (const float4*)(k + (size_t)(bstart + rel) * DSTATE);
        float acc = 0.f;
        for (int d4 = 0; d4 < DSTATE / 4; ++d4) {
            float4 qv = qsv[d4];
            float4 kv = krow[d4];
            acc += qv.x * kv.x + qv.y * kv.y + qv.z * kv.z + qv.w * kv.w;
        }
        sc[c] = acc * 0.0625f;       // 1/sqrt(256)
    }

    // top-8 via 8 wave-wide argmax rounds (ties -> lower index, like top_k)
    float ts[TOPK]; int tr[TOPK];
#pragma unroll
    for (int i = 0; i < TOPK; ++i) {
        float best = (sc[0] >= sc[1]) ? sc[0] : sc[1];
        int   bidx = (sc[0] >= sc[1]) ? 2 * lane : 2 * lane + 1;
        for (int off = 16; off; off >>= 1) {
            float ov = __shfl_xor(best, off);
            int   oi = __shfl_xor(bidx, off);
            if (ov > best || (ov == best && oi < bidx)) { best = ov; bidx = oi; }
        }
        ts[i] = best; tr[i] = bidx;
        if (bidx == 2 * lane)          sc[0] = NEG_INF;
        else if (bidx == 2 * lane + 1) sc[1] = NEG_INF;
    }

    // softmax over selected (ts[0] is the max)
    const float m  = ts[0];
    const float m0 = (m == NEG_INF) ? 0.f : m;
    float w[TOPK]; float Z = 0.f;
#pragma unroll
    for (int i = 0; i < TOPK; ++i) { w[i] = expf(ts[i] - m0); Z += w[i]; }
    const float invZ = 1.f / fmaxf(Z, 1e-30f);

    // weighted V gather: lane owns dims [lane*8, lane*8+8)
    const int dbase = lane * 8;
    float acc[8] = {0.f, 0.f, 0.f, 0.f, 0.f, 0.f, 0.f, 0.f};
#pragma unroll
    for (int i = 0; i < TOPK; ++i) {
        const float wi = w[i] * invZ;
        int rel = s_in_b - LOOKB + tr[i];
        rel = (rel < 0) ? 0 : rel;                      // matches jnp.clip; w==0 anyway
        const float* vrow = v + (size_t)(bstart + rel) * DSTATE + dbase;
#pragma unroll
        for (int j = 0; j < 8; ++j) acc[j] += wi * vrow[j];
    }
    __bf16* out = ui + (size_t)t * DMODEL + 3 * DSTATE + dbase;
#pragma unroll
    for (int j = 0; j < 8; ++j) out[j] = f2bf(acc[j]);
}

// ---------------------------------------------------------------------------
// Halt gate: sigmoid(h @ Wh + bh)[:,2]. One thread per token (tiny).
// ---------------------------------------------------------------------------
__global__ void gate_kernel(const float* __restrict__ h,
                            const float* __restrict__ Wh,
                            const float* __restrict__ bh,
                            float* __restrict__ gate) {
    int t = blockIdx.x * blockDim.x + threadIdx.x;
    if (t >= NTOK) return;
    float acc = bh[2];
    const float* hrow = h + (size_t)t * DSTATE;
    for (int d = 0; d < DSTATE; ++d) acc += hrow[d] * Wh[d * 3 + 2];
    gate[t] = 1.f / (1.f + expf(-acc));
}

// h += delta * gate ; refresh h_bf16 and ui block 0.
__global__ void update_h_kernel(float* __restrict__ h,
                                const float* __restrict__ delta,
                                const float* __restrict__ gate,
                                __bf16* __restrict__ h_bf,
                                __bf16* __restrict__ ui) {
    int idx = blockIdx.x * blockDim.x + threadIdx.x;
    int d = idx & (DSTATE - 1);
    int t = idx >> 8;
    float nh = h[idx] + delta[idx] * gate[t];
    h[idx] = nh;
    __bf16 b = f2bf(nh);
    h_bf[idx] = b;
    ui[(size_t)t * DMODEL + d] = b;
}

// ---------------------------------------------------------------------------
extern "C" void kernel_launch(void* const* d_in, const int* in_sizes, int n_in,
                              void* d_out, int out_size, void* d_ws, size_t ws_size,
                              hipStream_t stream) {
    (void)in_sizes; (void)n_in; (void)out_size; (void)ws_size;
    const float* x      = (const float*)d_in[0];
    const float* Wi     = (const float*)d_in[1];
    const float* bi     = (const float*)d_in[2];
    const float* conv_w = (const float*)d_in[3];
    const float* conv_b = (const float*)d_in[4];
    const float* Wp     = (const float*)d_in[5];
    const float* bp     = (const float*)d_in[6];
    const float* Wh     = (const float*)d_in[7];
    const float* bh     = (const float*)d_in[8];
    const float* Wq     = (const float*)d_in[9];
    const float* bq     = (const float*)d_in[10];
    const float* Wk     = (const float*)d_in[11];
    const float* bk     = (const float*)d_in[12];
    const float* Wv     = (const float*)d_in[13];
    const float* bv     = (const float*)d_in[14];
    const float* W1     = (const float*)d_in[15];
    const float* b1     = (const float*)d_in[16];
    const float* W2     = (const float*)d_in[17];
    const float* b2     = (const float*)d_in[18];
    const float* Wo     = (const float*)d_in[19];
    const float* bo     = (const float*)d_in[20];
    float* out = (float*)d_out;

    // --- workspace carve ---
    char* ws = (char*)d_ws;
    auto carve = [&](size_t bytes) -> void* {
        void* p = (void*)ws; ws += (bytes + 255) & ~(size_t)255; return p;
    };
    __bf16* x_bf   = (__bf16*)carve((size_t)NTOK * DMODEL * 2);
    float*  h      = (float*) carve((size_t)NTOK * DSTATE * 4);
    __bf16* h_bf   = (__bf16*)carve((size_t)NTOK * DSTATE * 2);
    __bf16* ui     = (__bf16*)carve((size_t)NTOK * DMODEL * 2);
    float*  qb     = (float*) carve((size_t)NTOK * DSTATE * 4);
    float*  kb     = (float*) carve((size_t)NTOK * DSTATE * 4);
    float*  vb     = (float*) carve((size_t)NTOK * DSTATE * 4);
    __bf16* t_bf   = (__bf16*)carve((size_t)NTOK * DSTATE * 2);
    float*  delta  = (float*) carve((size_t)NTOK * DSTATE * 4);
    float*  gate   = (float*) carve((size_t)NTOK * 4);
    __bf16* ph_bf  = (__bf16*)carve((size_t)NCHUNKR * DSTATE * 2);
    __bf16* pb_bf  = (__bf16*)carve((size_t)NCHUNKR * DSTATE * 2);
    __bf16* Wi_p   = (__bf16*)carve((size_t)DMODEL * DSTATE * 2);
    __bf16* Wq_p   = (__bf16*)carve((size_t)DSTATE * DSTATE * 2);
    __bf16* Wk_p   = (__bf16*)carve((size_t)DSTATE * DSTATE * 2);
    __bf16* Wv_p   = (__bf16*)carve((size_t)DSTATE * DSTATE * 2);
    __bf16* Wp_p   = (__bf16*)carve((size_t)DSTATE * DSTATE * 2);
    __bf16* W1_p   = (__bf16*)carve((size_t)(4 * DSTATE) * DSTATE * 2);
    __bf16* W2_p   = (__bf16*)carve((size_t)DSTATE * DSTATE * 2);
    __bf16* Wo_p   = (__bf16*)carve((size_t)DSTATE * DMODEL * 2);

    const int TB = 256;
    auto blocks = [](size_t n, int tb) { return (unsigned)((n + tb - 1) / tb); };

    // --- pack weights into WMMA B-fragment layout ---
    pack_w_kernel<<<blocks((size_t)DMODEL * DSTATE, TB), TB, 0, stream>>>(Wi, DMODEL, DSTATE, Wi_p);
    pack_w_kernel<<<blocks((size_t)DSTATE * DSTATE, TB), TB, 0, stream>>>(Wq, DSTATE, DSTATE, Wq_p);
    pack_w_kernel<<<blocks((size_t)DSTATE * DSTATE, TB), TB, 0, stream>>>(Wk, DSTATE, DSTATE, Wk_p);
    pack_w_kernel<<<blocks((size_t)DSTATE * DSTATE, TB), TB, 0, stream>>>(Wv, DSTATE, DSTATE, Wv_p);
    pack_w_kernel<<<blocks((size_t)DSTATE * DSTATE, TB), TB, 0, stream>>>(Wp, DSTATE, DSTATE, Wp_p);
    pack_w_kernel<<<blocks((size_t)(4 * DSTATE) * DSTATE, TB), TB, 0, stream>>>(W1, 4 * DSTATE, DSTATE, W1_p);
    pack_w_kernel<<<blocks((size_t)DSTATE * DSTATE, TB), TB, 0, stream>>>(W2, DSTATE, DSTATE, W2_p);
    pack_w_kernel<<<blocks((size_t)DSTATE * DMODEL, TB), TB, 0, stream>>>(Wo, DSTATE, DMODEL, Wo_p);

    // --- x -> bf16, then h = x @ Wi + bi ---
    convert_bf16_kernel<<<blocks((size_t)NTOK * DMODEL, TB), TB, 0, stream>>>(x, x_bf, NTOK * DMODEL);
    gemm_bf16_wmma_kernel<<<dim3(NTOK / 256, DSTATE / 64), TB, 0, stream>>>(
        x_bf, Wi_p, bi, h, DSTATE, nullptr, 0, nullptr, 0, NTOK, DSTATE, DMODEL, 0);
    seed_h_kernel<<<blocks((size_t)NTOK * DSTATE, TB), TB, 0, stream>>>(h, h_bf, ui);

    for (int step = 0; step < STEPS; ++step) {
        // ui block 1: causal depthwise conv
        conv_kernel<<<blocks((size_t)NTOK * DSTATE, TB), TB, 0, stream>>>(h, conv_w, conv_b, ui);
        // ui block 2: phrase pool -> Wp -> broadcast
        chunk_mean_kernel<<<blocks((size_t)NCHUNKR * DSTATE, TB), TB, 0, stream>>>(h, ph_bf);
        gemm_bf16_wmma_kernel<<<dim3(NCHUNKR / 256, DSTATE / 64), TB, 0, stream>>>(
            ph_bf, Wp_p, bp, nullptr, 0, pb_bf, DSTATE, nullptr, 0, NCHUNKR, DSTATE, DSTATE, 0);
        broadcast_pb_kernel<<<blocks((size_t)NTOK * DSTATE, TB), TB, 0, stream>>>(pb_bf, ui);
        // q, k, v projections
        gemm_bf16_wmma_kernel<<<dim3(NTOK / 256, DSTATE / 64), TB, 0, stream>>>(
            h_bf, Wq_p, bq, qb, DSTATE, nullptr, 0, nullptr, 0, NTOK, DSTATE, DSTATE, 0);
        gemm_bf16_wmma_kernel<<<dim3(NTOK / 256, DSTATE / 64), TB, 0, stream>>>(
            h_bf, Wk_p, bk, kb, DSTATE, nullptr, 0, nullptr, 0, NTOK, DSTATE, DSTATE, 0);
        gemm_bf16_wmma_kernel<<<dim3(NTOK / 256, DSTATE / 64), TB, 0, stream>>>(
            h_bf, Wv_p, bv, vb, DSTATE, nullptr, 0, nullptr, 0, NTOK, DSTATE, DSTATE, 0);
        // ui block 3: windowed causal top-8 attention
        attn_topk_kernel<<<NTOK / 8, TB, 0, stream>>>(qb, kb, vb, ui);
        // halt gate
        gate_kernel<<<blocks(NTOK, TB), TB, 0, stream>>>(h, Wh, bh, gate);
        // delta = tanh(ui @ W1 + b1) @ W2 + b2
        gemm_bf16_wmma_kernel<<<dim3(NTOK / 256, DSTATE / 64), TB, 0, stream>>>(
            ui, W1_p, b1, nullptr, 0, t_bf, DSTATE, nullptr, 0, NTOK, DSTATE, DMODEL, 1);
        gemm_bf16_wmma_kernel<<<dim3(NTOK / 256, DSTATE / 64), TB, 0, stream>>>(
            t_bf, W2_p, b2, delta, DSTATE, nullptr, 0, nullptr, 0, NTOK, DSTATE, DSTATE, 0);
        // h += delta * gate
        update_h_kernel<<<blocks((size_t)NTOK * DSTATE, TB), TB, 0, stream>>>(h, delta, gate, h_bf, ui);
    }

    // out = x + h @ Wo + bo
    gemm_bf16_wmma_kernel<<<dim3(NTOK / 256, DMODEL / 64), TB, 0, stream>>>(
        h_bf, Wo_p, bo, out, DMODEL, nullptr, 0, x, DMODEL, NTOK, DMODEL, DSTATE, 0);
}
